// ACEncoder_44220983280251
// MI455X (gfx1250) — matrix operand order repaired and checked by reading, not compiled, not run on hardware
//
#include <hip/hip_runtime.h>
#include <math.h>

#define EMB 128
#define NGRAPH 64

typedef __attribute__((ext_vector_type(16))) __bf16 v16bf;
typedef __attribute__((ext_vector_type(8)))  float  v8f;

// ---------------------------------------------------------------- utilities
__global__ void fill_f32_k(float* __restrict__ p, float v, int n) {
  int i = blockIdx.x * blockDim.x + threadIdx.x;
  if (i < n) p[i] = v;
}

// pad (N,8) -> (N,32), zero-filled
__global__ void pad8to32_k(const float* __restrict__ x, float* __restrict__ xp, int n32) {
  int i = blockIdx.x * blockDim.x + threadIdx.x;
  if (i < n32) {
    int f = i & 31;
    xp[i] = (f < 8) ? x[(i >> 5) * 8 + f] : 0.f;
  }
}

// wt[f,k] = w[k,f]  for (128,128)
__global__ void transpose128_k(const float* __restrict__ w, float* __restrict__ wt) {
  int i = blockIdx.x * blockDim.x + threadIdx.x;
  if (i < 128 * 128) {
    int f = i >> 7, k = i & 127;
    wt[i] = w[k * 128 + f];
  }
}

// ---------------------------------------------------------------- WMMA GEMM
__device__ __forceinline__ void pack4(v16bf& v, int base, float4 f) {
  v[base + 0] = (__bf16)f.x;
  v[base + 1] = (__bf16)f.y;
  v[base + 2] = (__bf16)f.z;
  v[base + 3] = (__bf16)f.w;
}

// Y[M,128] = X[M,K] @ W^T  with W stored (128, K), K compile-time (32 or 128).
// bias optional (len 128); act: 0 = none, 1 = tanh.
// block = 256 (8 waves); wave w covers col tile w*16; block covers 32 rows (2 tiles/wave).
template <int K>
__global__ void gemm_wmma_bf16_k(const float* __restrict__ X,
                                 const float* __restrict__ W,
                                 const float* __restrict__ bias,
                                 float* __restrict__ Y,
                                 int M, int act) {
  const int F = 128;
  const int lane = threadIdx.x & 31;
  const int wave = threadIdx.x >> 5;             // 0..7
  const int lm = lane & 15;
  const int lh = lane >> 4;
  const int col = wave * 16 + lm;

  const int row0 = blockIdx.x * 32 + lm;         // tile 0
  const int row1 = row0 + 16;                    // tile 1
  const int rc0 = row0 < M ? row0 : M - 1;       // clamp: loads always in-bounds
  const int rc1 = row1 < M ? row1 : M - 1;
  const float* __restrict__ Xr0 = X + (size_t)rc0 * K;
  const float* __restrict__ Xr1 = X + (size_t)rc1 * K;
  const float* __restrict__ Wr  = W + (size_t)col * K;

  v8f acc0 = {0.f, 0.f, 0.f, 0.f, 0.f, 0.f, 0.f, 0.f};
  v8f acc1 = {0.f, 0.f, 0.f, 0.f, 0.f, 0.f, 0.f, 0.f};

#pragma unroll
  for (int kb = 0; kb < K; kb += 32) {
    if (kb + 32 < K) {                           // compile-time folded
      __builtin_prefetch(Xr0 + kb + 32, 0, 1);
      __builtin_prefetch(Xr1 + kb + 32, 0, 1);
    }
    // A fragment (16x32 bf16): elem i -> K_rel = (i/8)*16 + lh*8 + (i%8)
    v16bf a0, a1, b;
    pack4(a0, 0,  *(const float4*)(Xr0 + kb + lh * 8));
    pack4(a0, 4,  *(const float4*)(Xr0 + kb + lh * 8 + 4));
    pack4(a0, 8,  *(const float4*)(Xr0 + kb + 16 + lh * 8));
    pack4(a0, 12, *(const float4*)(Xr0 + kb + 16 + lh * 8 + 4));
    pack4(a1, 0,  *(const float4*)(Xr1 + kb + lh * 8));
    pack4(a1, 4,  *(const float4*)(Xr1 + kb + lh * 8 + 4));
    pack4(a1, 8,  *(const float4*)(Xr1 + kb + 16 + lh * 8));
    pack4(a1, 12, *(const float4*)(Xr1 + kb + 16 + lh * 8 + 4));
    // B fragment (32x16 bf16): elem i -> K_rel = (i/8)*16 + lh*8 + (i%8)
    pack4(b, 0,  *(const float4*)(Wr + kb + lh * 8));
    pack4(b, 4,  *(const float4*)(Wr + kb + lh * 8 + 4));
    pack4(b, 8,  *(const float4*)(Wr + kb + 16 + lh * 8));
    pack4(b, 12, *(const float4*)(Wr + kb + 16 + lh * 8 + 4));

    acc0 = __builtin_amdgcn_wmma_f32_16x16x32_bf16(false, a0, false, b,
                                                   (short)0, acc0, false, false);
    acc1 = __builtin_amdgcn_wmma_f32_16x16x32_bf16(false, a1, false, b,
                                                   (short)0, acc1, false, false);
  }

  float bb = bias ? bias[col] : 0.f;
#pragma unroll
  for (int r = 0; r < 8; ++r) {
    int m0 = blockIdx.x * 32 + (lh << 3) + r;    // C/D: VGPR r -> M = r + 8*(lane/16)
    if (m0 < M) {
      float o = acc0[r] + bb;
      if (act == 1) o = tanhf(o);
      Y[(size_t)m0 * F + col] = o;
    }
    int m1 = m0 + 16;
    if (m1 < M) {
      float o = acc1[r] + bb;
      if (act == 1) o = tanhf(o);
      Y[(size_t)m1 * F + col] = o;
    }
  }
}

// ---------------------------------------------------------------- GAT pieces
// wa[d] = sum_f We[f,d] * att[f]   (We is (128,4))
__global__ void compute_wa_k(const float* __restrict__ We, const float* __restrict__ att,
                             float* __restrict__ wa) {
  __shared__ float s[4];
  int f = threadIdx.x;                           // 128 threads
  if (f < 4) s[f] = 0.f;
  __syncthreads();
  float a = att[f];
#pragma unroll
  for (int d = 0; d < 4; ++d) atomicAdd(&s[d], We[f * 4 + d] * a);
  __syncthreads();
  if (f < 4) wa[f] = s[f];
}

// per-node attention scalars: salpha = x_src . att[:128], dalpha = x_dst . att[128:]
__global__ void node_alpha_k(const float* __restrict__ xs, const float* __restrict__ xd,
                             const float* __restrict__ att,
                             float* __restrict__ salpha, float* __restrict__ dalpha, int N) {
  int lane = threadIdx.x & 31;
  int node = (blockIdx.x * blockDim.x + threadIdx.x) >> 5;
  if (node >= N) return;
  const float4* xs4 = (const float4*)(xs + (size_t)node * EMB);
  const float4* xd4 = (const float4*)(xd + (size_t)node * EMB);
  const float4* a14 = (const float4*)att;
  const float4* a24 = (const float4*)(att + EMB);
  float4 xv = xs4[lane], av = a14[lane];
  float s = xv.x * av.x + xv.y * av.y + xv.z * av.z + xv.w * av.w;
  xv = xd4[lane]; av = a24[lane];
  float d = xv.x * av.x + xv.y * av.y + xv.z * av.z + xv.w * av.w;
#pragma unroll
  for (int m = 16; m >= 1; m >>= 1) {
    s += __shfl_xor(s, m, 32);
    d += __shfl_xor(d, m, 32);
  }
  if (lane == 0) { salpha[node] = s; dalpha[node] = d; }
}

__global__ void edge_logit_k(const int* __restrict__ src, const int* __restrict__ dst,
                             const float* __restrict__ ea, const float* __restrict__ wa,
                             const float* __restrict__ salpha, const float* __restrict__ dalpha,
                             float* __restrict__ ealpha, int E) {
  int e = blockIdx.x * blockDim.x + threadIdx.x;
  if (e >= E) return;
  float4 eav = ((const float4*)ea)[e];
  float v = salpha[src[e]] + dalpha[dst[e]]
          + eav.x * wa[0] + eav.y * wa[1] + eav.z * wa[2] + eav.w * wa[3];
  ealpha[e] = v > 0.f ? v : 0.2f * v;            // leaky_relu 0.2
}

__global__ void block_max_k(const float* __restrict__ x, int n, float* __restrict__ out) {
  __shared__ float sm[256];
  float m = -3.4e38f;
  for (int i = blockIdx.x * blockDim.x + threadIdx.x; i < n; i += gridDim.x * blockDim.x)
    m = fmaxf(m, x[i]);
  sm[threadIdx.x] = m;
  __syncthreads();
  for (int s = 128; s >= 1; s >>= 1) {
    if ((int)threadIdx.x < s) sm[threadIdx.x] = fmaxf(sm[threadIdx.x], sm[threadIdx.x + s]);
    __syncthreads();
  }
  if (threadIdx.x == 0) out[blockIdx.x] = sm[0];
}

__global__ void edge_exp_k(float* __restrict__ ealpha, const float* __restrict__ gmax,
                           float* __restrict__ denom, const int* __restrict__ dst, int E) {
  int e = blockIdx.x * blockDim.x + threadIdx.x;
  if (e >= E) return;
  float v = expf(ealpha[e] - gmax[0]);
  ealpha[e] = v;
  atomicAdd(&denom[dst[e]], v);
}

__global__ void init_bias_k(float* __restrict__ out, const float* __restrict__ bias, int N) {
  int i = blockIdx.x * blockDim.x + threadIdx.x;
  if (i < N * EMB) out[i] = bias[i & (EMB - 1)];
}

// wave per edge: out[dst] += x_src[src] * (e / (denom[dst]+eps))
__global__ void edge_msg_k(const int* __restrict__ src, const int* __restrict__ dst,
                           const float* __restrict__ ealpha, const float* __restrict__ denom,
                           const float* __restrict__ xsrc, float* __restrict__ out, int E) {
  int lane = threadIdx.x & 31;
  int e = (blockIdx.x * blockDim.x + threadIdx.x) >> 5;
  if (e >= E) return;
  int s = src[e], d = dst[e];
  float a = ealpha[e] / (denom[d] + 1e-12f);
  float4 xv = ((const float4*)(xsrc + (size_t)s * EMB))[lane];
  float* orow = out + (size_t)d * EMB + lane * 4;
  atomicAdd(orow + 0, xv.x * a);
  atomicAdd(orow + 1, xv.y * a);
  atomicAdd(orow + 2, xv.z * a);
  atomicAdd(orow + 3, xv.w * a);
}

__global__ void combine1_k(float* __restrict__ h, const float* __restrict__ c,
                           const float* __restrict__ x0, int n) {
  int i = blockIdx.x * blockDim.x + threadIdx.x;
  if (i < n) h[i] = 0.8f * c[i] + 0.2f * x0[i];
}

__global__ void combine2_k(float* __restrict__ h, const float* __restrict__ t, float beta, int n) {
  int i = blockIdx.x * blockDim.x + threadIdx.x;
  if (i < n) {
    float v = (1.f - beta) * h[i] + beta * t[i];
    h[i] = v > 0.f ? v : 0.01f * v;              // leaky_relu 0.01
  }
}

// ---------------------------------------------------------------- pooling
__global__ void pool_sum_k(const float* __restrict__ x, const int* __restrict__ batch,
                           float* __restrict__ s, float* __restrict__ cnt, int N) {
  int lane = threadIdx.x & 31;
  int node = (blockIdx.x * blockDim.x + threadIdx.x) >> 5;
  if (node >= N) return;
  int g = batch[node];
  float4 xv = ((const float4*)(x + (size_t)node * EMB))[lane];
  float* srow = s + g * EMB + lane * 4;
  atomicAdd(srow + 0, xv.x);
  atomicAdd(srow + 1, xv.y);
  atomicAdd(srow + 2, xv.z);
  atomicAdd(srow + 3, xv.w);
  if (lane == 0) atomicAdd(&cnt[g], 1.f);
}

__global__ void pool_mean_k(const float* __restrict__ s, const float* __restrict__ cnt,
                            float* __restrict__ mean) {
  int i = blockIdx.x * blockDim.x + threadIdx.x;
  if (i < NGRAPH * EMB) mean[i] = s[i] / fmaxf(cnt[i >> 7], 1.f);
}

__global__ void pool_gap_k(const float* __restrict__ x, const int* __restrict__ batch,
                           const float* __restrict__ t, float* __restrict__ gap, int N) {
  int lane = threadIdx.x & 31;
  int node = (blockIdx.x * blockDim.x + threadIdx.x) >> 5;
  if (node >= N) return;
  int g = batch[node];
  float4 xv = ((const float4*)(x + (size_t)node * EMB))[lane];
  float4 tv = ((const float4*)(t + g * EMB))[lane];
  float dotv = xv.x * tv.x + xv.y * tv.y + xv.z * tv.z + xv.w * tv.w;
#pragma unroll
  for (int m = 16; m >= 1; m >>= 1) dotv += __shfl_xor(dotv, m, 32);
  float cf = 1.f / (1.f + expf(-10.f * dotv));
  float* grow = gap + g * EMB + lane * 4;
  atomicAdd(grow + 0, cf * xv.x);
  atomicAdd(grow + 1, cf * xv.y);
  atomicAdd(grow + 2, cf * xv.z);
  atomicAdd(grow + 3, cf * xv.w);
}

__global__ void pool_out_k(const float* __restrict__ gap, const float* __restrict__ mean,
                           const float* __restrict__ s, float* __restrict__ out, int base) {
  int i = blockIdx.x * blockDim.x + threadIdx.x;
  if (i < NGRAPH * EMB) {
    int g = i >> 7, f = i & 127;
    out[g * 2 * EMB + base + f] = gap[i] + mean[i] + s[i];
  }
}

// ---------------------------------------------------------------- host side
namespace {

struct ConvP {
  const float* Wdst; const float* We; const float* Wsrc; const float* att; const float* bias;
};

struct Scratch {
  float *xsrc, *xdst, *x0, *h, *c, *tmp;
  float *xpad, *wpad, *wt;
  float *salpha, *dalpha, *denom, *ealpha;
  float *wa, *red, *gmax;
  float *s, *mean, *t, *gap, *cnt;
};

inline int cdiv(int a, int b) { return (a + b - 1) / b; }

void gemm128(const float* X, const float* W, const float* bias, float* Y,
             int M, int act, hipStream_t st) {
  gemm_wmma_bf16_k<128><<<dim3(cdiv(M, 32)), dim3(256), 0, st>>>(X, W, bias, Y, M, act);
}
void gemm32(const float* X, const float* W, const float* bias, float* Y,
            int M, int act, hipStream_t st) {
  gemm_wmma_bf16_k<32><<<dim3(cdiv(M, 32)), dim3(256), 0, st>>>(X, W, bias, Y, M, act);
}

void run_gat_conv(const float* x, int N, int Fin,
                  const int* src, const int* dst, const float* ea, int E,
                  const ConvP& cp, float* out, Scratch& ws, hipStream_t st) {
  if (Fin == 8) {
    // zero-pad activations and the two weight matrices to K=32 for branch-free WMMA
    pad8to32_k<<<cdiv(N * 32, 256), 256, 0, st>>>(x, ws.xpad, N * 32);
    pad8to32_k<<<cdiv(128 * 32, 256), 256, 0, st>>>(cp.Wsrc, ws.wpad, 128 * 32);
    gemm32(ws.xpad, ws.wpad, nullptr, ws.xsrc, N, 0, st);
    pad8to32_k<<<cdiv(128 * 32, 256), 256, 0, st>>>(cp.Wdst, ws.wpad, 128 * 32);
    gemm32(ws.xpad, ws.wpad, nullptr, ws.xdst, N, 0, st);
  } else {
    gemm128(x, cp.Wsrc, nullptr, ws.xsrc, N, 0, st);
    gemm128(x, cp.Wdst, nullptr, ws.xdst, N, 0, st);
  }
  compute_wa_k<<<1, 128, 0, st>>>(cp.We, cp.att, ws.wa);
  node_alpha_k<<<cdiv(N * 32, 256), 256, 0, st>>>(ws.xsrc, ws.xdst, cp.att,
                                                  ws.salpha, ws.dalpha, N);
  edge_logit_k<<<cdiv(E, 256), 256, 0, st>>>(src, dst, ea, ws.wa,
                                             ws.salpha, ws.dalpha, ws.ealpha, E);
  block_max_k<<<256, 256, 0, st>>>(ws.ealpha, E, ws.red);
  block_max_k<<<1, 256, 0, st>>>(ws.red, 256, ws.gmax);
  fill_f32_k<<<cdiv(N, 256), 256, 0, st>>>(ws.denom, 0.f, N);
  edge_exp_k<<<cdiv(E, 256), 256, 0, st>>>(ws.ealpha, ws.gmax, ws.denom, dst, E);
  init_bias_k<<<cdiv(N * EMB, 256), 256, 0, st>>>(out, cp.bias, N);
  edge_msg_k<<<cdiv(E * 32, 256), 256, 0, st>>>(src, dst, ws.ealpha, ws.denom,
                                                ws.xsrc, out, E);
}

void run_deep_gat(const float* x, int N, int Fin,
                  const int* src, const int* dst, const float* ea, int E,
                  const ConvP* convs_mid, const float* const* wmid, int num_mid,
                  const ConvP& conv_s, const ConvP& conv_e,
                  float* emb_out, Scratch& ws, hipStream_t st) {
  run_gat_conv(x, N, Fin, src, dst, ea, E, conv_s, ws.x0, ws, st);
  float* hcur = ws.x0;
  int n = N * EMB;
  for (int i = 0; i < num_mid; ++i) {
    run_gat_conv(hcur, N, EMB, src, dst, ea, E, convs_mid[i], ws.c, ws, st);
    combine1_k<<<cdiv(n, 256), 256, 0, st>>>(ws.h, ws.c, ws.x0, n);
    transpose128_k<<<cdiv(128 * 128, 256), 256, 0, st>>>(wmid[i], ws.wt);
    gemm128(ws.h, ws.wt, nullptr, ws.tmp, N, 0, st);
    float beta = logf(0.2f / (float)(i + 1) + 1.f);
    combine2_k<<<cdiv(n, 256), 256, 0, st>>>(ws.h, ws.tmp, beta, n);
    hcur = ws.h;
  }
  run_gat_conv(hcur, N, EMB, src, dst, ea, E, conv_e, emb_out, ws, st);
}

void run_pool(const float* emb, const int* batch, int N, const float* gap_w,
              float* out, int base, Scratch& ws, hipStream_t st) {
  fill_f32_k<<<cdiv(NGRAPH * EMB, 256), 256, 0, st>>>(ws.s, 0.f, NGRAPH * EMB);
  fill_f32_k<<<cdiv(NGRAPH * EMB, 256), 256, 0, st>>>(ws.gap, 0.f, NGRAPH * EMB);
  fill_f32_k<<<1, 256, 0, st>>>(ws.cnt, 0.f, NGRAPH);
  pool_sum_k<<<cdiv(N * 32, 256), 256, 0, st>>>(emb, batch, ws.s, ws.cnt, N);
  pool_mean_k<<<cdiv(NGRAPH * EMB, 256), 256, 0, st>>>(ws.s, ws.cnt, ws.mean);
  transpose128_k<<<cdiv(128 * 128, 256), 256, 0, st>>>(gap_w, ws.wt);
  gemm128(ws.mean, ws.wt, nullptr, ws.t, NGRAPH, /*act=tanh*/1, st);
  pool_gap_k<<<cdiv(N * 32, 256), 256, 0, st>>>(emb, batch, ws.t, ws.gap, N);
  pool_out_k<<<cdiv(NGRAPH * EMB, 256), 256, 0, st>>>(ws.gap, ws.mean, ws.s, out, base);
}

} // namespace

extern "C" void kernel_launch(void* const* d_in, const int* in_sizes, int n_in,
                              void* d_out, int out_size, void* d_ws, size_t ws_size,
                              hipStream_t stream) {
  (void)n_in; (void)out_size; (void)ws_size;
  const float* p_x  = (const float*)d_in[0];
  const int*   p_ei = (const int*)d_in[1];
  const float* p_ea = (const float*)d_in[2];
  const int*   p_b  = (const int*)d_in[3];
  const float* v_x  = (const float*)d_in[4];
  const int*   v_ei = (const int*)d_in[5];
  const float* v_ea = (const float*)d_in[6];
  const int*   v_b  = (const int*)d_in[7];

  const int Np = in_sizes[0] / 8;
  const int Ep = in_sizes[1] / 2;
  const int Nv = in_sizes[4] / 8;
  const int Ev = in_sizes[5] / 2;

  // pytree leaves in jax sorted-key order:
  // p_params: conv_0, conv_1, conv_2, conv_e, conv_s (each: Wdst, We, Wsrc, att, bias), w_0..w_2
  auto conv_at = [&](int b) -> ConvP {
    return ConvP{(const float*)d_in[b], (const float*)d_in[b + 1], (const float*)d_in[b + 2],
                 (const float*)d_in[b + 3], (const float*)d_in[b + 4]};
  };
  ConvP p_mid[3] = {conv_at(8), conv_at(13), conv_at(18)};
  ConvP p_conv_e = conv_at(23), p_conv_s = conv_at(28);
  const float* p_w[3] = {(const float*)d_in[33], (const float*)d_in[34], (const float*)d_in[35]};
  ConvP v_mid[1] = {conv_at(36)};
  ConvP v_conv_e = conv_at(41), v_conv_s = conv_at(46);
  const float* v_w[1] = {(const float*)d_in[51]};
  const float* p_gap_w = (const float*)d_in[52];
  const float* v_gap_w = (const float*)d_in[53];

  // ---- workspace carve-out (p-graph sizes are the maxima; v reuses the same buffers)
  char* cur = (char*)d_ws;
  auto take = [&](size_t bytes) -> float* {
    float* p = (float*)cur;
    cur += (bytes + 255) & ~(size_t)255;
    return p;
  };
  const size_t nodeB = (size_t)Np * EMB * sizeof(float);
  Scratch ws;
  ws.xsrc = take(nodeB);  ws.xdst = take(nodeB);
  ws.x0   = take(nodeB);  ws.h    = take(nodeB);
  ws.c    = take(nodeB);  ws.tmp  = take(nodeB);
  ws.xpad = take((size_t)Np * 32 * sizeof(float));
  ws.wpad = take((size_t)128 * 32 * sizeof(float));
  ws.wt   = take((size_t)128 * 128 * sizeof(float));
  ws.salpha = take((size_t)Np * sizeof(float));
  ws.dalpha = take((size_t)Np * sizeof(float));
  ws.denom  = take((size_t)Np * sizeof(float));
  ws.ealpha = take((size_t)Ep * sizeof(float));
  ws.wa   = take(16 * sizeof(float));
  ws.red  = take(256 * sizeof(float));
  ws.gmax = take(16 * sizeof(float));
  ws.s    = take((size_t)NGRAPH * EMB * sizeof(float));
  ws.mean = take((size_t)NGRAPH * EMB * sizeof(float));
  ws.t    = take((size_t)NGRAPH * EMB * sizeof(float));
  ws.gap  = take((size_t)NGRAPH * EMB * sizeof(float));
  ws.cnt  = take((size_t)NGRAPH * sizeof(float));

  float* out = (float*)d_out;

  // ---- p graph: deep GAT (num_mid=3) + pooling into out[:, 0:128]
  run_deep_gat(p_x, Np, 8, p_ei, p_ei + Ep, p_ea,
               Ep, p_mid, p_w, 3, p_conv_s, p_conv_e, ws.c, ws, stream);
  run_pool(ws.c, p_b, Np, p_gap_w, out, /*base=*/0, ws, stream);

  // ---- v graph: deep GAT (num_mid=1) + pooling into out[:, 128:256]
  run_deep_gat(v_x, Nv, 8, v_ei, v_ei + Ev, v_ea,
               Ev, v_mid, v_w, 1, v_conv_s, v_conv_e, ws.c, ws, stream);
  run_pool(ws.c, v_b, Nv, v_gap_w, out, /*base=*/EMB, ws, stream);
}
